// TransformerEncoder_9534827397386
// MI455X (gfx1250) — matrix operand order repaired and checked by reading, not compiled, not run on hardware
//
#include <hip/hip_runtime.h>

// ---------------------------------------------------------------------------
// CDNA5 (gfx1250) transformer encoder forward pass.
//  - bf16 WMMA (v_wmma_f32_16x16x32_bf16) for all GEMMs + flash attention
//  - bf16 activation streams (halves HBM traffic on an HBM-bound pass)
//  - A-fragments fed straight from global memory (b128 loads, no LDS stage)
//  - GLOBAL_LOAD_ASYNC_TO_LDS_B128 + s_wait_asynccnt for attention Q/K tiles
// ---------------------------------------------------------------------------

typedef __attribute__((ext_vector_type(16))) __bf16 v16bf;
typedef __attribute__((ext_vector_type(8)))  float  v8f;
typedef unsigned short bf16_t;

union Frag16 { v16bf v; unsigned int u[8]; uint4 q[2]; };

static __device__ __forceinline__ bf16_t f2bf(float f) {
  union { float f; unsigned int u; } c; c.f = f;
  return (bf16_t)((c.u + 0x7FFFu + ((c.u >> 16) & 1u)) >> 16);
}

static __device__ __forceinline__ v8f wmma_bf16(const v16bf& a, const v16bf& b, v8f c) {
  return __builtin_amdgcn_wmma_f32_16x16x32_bf16(false, a, false, b, (short)0, c,
                                                 false, false);
}

// CDNA5 async global->LDS copy (16B per lane), GVS addressing:
//   vdst = LDS byte address (VGPR), vaddr = 32-bit byte offset, saddr = base.
static __device__ __forceinline__ void async_copy_b128(unsigned lds_addr,
                                                       unsigned voff,
                                                       const void* sbase) {
  asm volatile("global_load_async_to_lds_b128 %0, %1, %2"
               :
               : "v"(lds_addr), "v"(voff), "s"(sbase)
               : "memory");
}
static __device__ __forceinline__ void wait_async0() {
  asm volatile("s_wait_asynccnt 0x0" ::: "memory");
}

// ---------------------------------------------------------------------------
// LayerNorm: one wave32 per 768-element row, bf16 output.
// ---------------------------------------------------------------------------
__global__ __launch_bounds__(256) void layernorm_kernel(
    const float* __restrict__ X, const float* __restrict__ w,
    const float* __restrict__ b, bf16_t* __restrict__ Y) {
  const int lane = threadIdx.x & 31;
  const int wave = threadIdx.x >> 5;
  const int row  = blockIdx.x * 8 + wave;
  const float* x = X + (size_t)row * 768;

  float v[24];
  float s = 0.0f;
#pragma unroll
  for (int i = 0; i < 24; ++i) { v[i] = x[lane + i * 32]; s += v[i]; }
#pragma unroll
  for (int m = 16; m >= 1; m >>= 1) s += __shfl_xor(s, m, 32);
  const float mu = s * (1.0f / 768.0f);

  float var = 0.0f;
#pragma unroll
  for (int i = 0; i < 24; ++i) { float d = v[i] - mu; var += d * d; }
#pragma unroll
  for (int m = 16; m >= 1; m >>= 1) var += __shfl_xor(var, m, 32);
  const float rstd = rsqrtf(var * (1.0f / 768.0f) + 1e-6f);

  bf16_t* y = Y + (size_t)row * 768;
#pragma unroll
  for (int i = 0; i < 24; ++i) {
    int c = lane + i * 32;
    y[c] = f2bf((v[i] - mu) * rstd * w[c] + b[c]);
  }
}

// ---------------------------------------------------------------------------
// GEMM: C[M,Nc] = act(A_bf16[M,K] @ W_f32[K,Nc] + bias) (+ resid_f32)
// 128x128 block tile, BK=32, 8 waves each computing 32(M)x64(N).
// A-fragments: two global b128 loads per frag (bf16 row-major == A layout).
// W: float4 loads, f32->bf16, transposed into LDS so B-frags are contiguous.
// flags: bit0 = exact GELU, bit1 = bf16 output (else f32 + optional resid).
// ---------------------------------------------------------------------------
#define GBM 128
#define GBN 128
#define GBK 32
#define LDB_S 40  // halves per transposed W row in LDS (even -> dword aligned)

__global__ __launch_bounds__(256) void gemm_kernel(
    const bf16_t* __restrict__ A, const float* __restrict__ W,
    const float* __restrict__ bias, const float* __restrict__ resid,
    void* __restrict__ Cout, int M, int K, int Nc, int flags) {
  __shared__ bf16_t Bs[GBN * LDB_S];  // [n][k]

  const int tid  = threadIdx.x;
  const int lane = tid & 31;
  const int wave = tid >> 5;
  const int lo   = lane & 15;
  const int hi   = lane >> 4;

  const int blockM = blockIdx.y * GBM;
  const int blockN = blockIdx.x * GBN;
  const int waveM  = (wave & 3) * 32;
  const int waveN  = (wave >> 2) * 64;

  v8f acc[2][4];
#pragma unroll
  for (int i = 0; i < 2; ++i)
#pragma unroll
    for (int j = 0; j < 4; ++j) acc[i][j] = {0.f, 0.f, 0.f, 0.f, 0.f, 0.f, 0.f, 0.f};

  for (int k0 = 0; k0 < K; k0 += GBK) {
    __syncthreads();
#pragma unroll
    for (int i = 0; i < 4; ++i) {                // 32k x 128n W tile, transposed
      int u  = tid + i * 256;                    // 0..1023
      int kk = u & 31, n4 = u >> 5;              // n4: 0..31
      const float4 w4 =
          *(const float4*)(W + (size_t)(k0 + kk) * Nc + blockN + n4 * 4);
      Bs[(n4 * 4 + 0) * LDB_S + kk] = f2bf(w4.x);
      Bs[(n4 * 4 + 1) * LDB_S + kk] = f2bf(w4.y);
      Bs[(n4 * 4 + 2) * LDB_S + kk] = f2bf(w4.z);
      Bs[(n4 * 4 + 3) * LDB_S + kk] = f2bf(w4.w);
    }
    __syncthreads();

    Frag16 af[2], bf[4];
#pragma unroll
    for (int mi = 0; mi < 2; ++mi) {             // A direct from global (bf16)
      const bf16_t* arow =
          A + (size_t)(blockM + waveM + mi * 16 + lo) * K + k0;
      af[mi].q[0] = *(const uint4*)(arow + 16 * hi);       // K = 0..7  (+16*hi)
      af[mi].q[1] = *(const uint4*)(arow + 16 * hi + 8);   // K = 8..15 (+16*hi)
    }
#pragma unroll
    for (int ni = 0; ni < 4; ++ni) {
      const unsigned int* col =
          (const unsigned int*)&Bs[(waveN + ni * 16 + lo) * LDB_S];
#pragma unroll
      for (int e = 0; e < 8; ++e)                // B layout: K = e + 16*hi
        bf[ni].u[e] = col[8 * hi + e];
    }
#pragma unroll
    for (int mi = 0; mi < 2; ++mi)
#pragma unroll
      for (int ni = 0; ni < 4; ++ni)
        acc[mi][ni] = wmma_bf16(af[mi].v, bf[ni].v, acc[mi][ni]);
  }

  float* Cf  = (float*)Cout;
  bf16_t* Ch = (bf16_t*)Cout;
#pragma unroll
  for (int mi = 0; mi < 2; ++mi)
#pragma unroll
    for (int ni = 0; ni < 4; ++ni) {
      const int col = blockN + waveN + ni * 16 + lo;
      const float bi = bias[col];
#pragma unroll
      for (int r = 0; r < 8; ++r) {
        const int row = blockM + waveM + mi * 16 + r + 8 * hi;
        const size_t idx = (size_t)row * Nc + col;
        float v = acc[mi][ni][r] + bi;
        if (flags & 1) v = 0.5f * v * (1.0f + erff(v * 0.70710678118654752f));
        if (flags & 2) {
          Ch[idx] = f2bf(v);
        } else {
          if (resid) v += resid[idx];
          Cf[idx] = v;
        }
      }
    }
}

// ---------------------------------------------------------------------------
// Flash attention over bf16 QKV: one block per (b, h, 128-query tile).
// Q/K tiles staged with GLOBAL_LOAD_ASYNC_TO_LDS_B128; V transposed manually;
// online softmax; P routed through per-wave LDS back into A layout.
// ---------------------------------------------------------------------------
#define NQT 128
#define NKC 64
#define DH  64
#define QS_S 72
#define KS_S 72
#define VS_S 72
#define PS_S 72

__global__ __launch_bounds__(256) void attn_kernel(
    const bf16_t* __restrict__ QKV, bf16_t* __restrict__ O, int B, int N, int H) {
  __shared__ alignas(16) bf16_t Qs[NQT * QS_S];
  __shared__ alignas(16) bf16_t Ks[NKC * KS_S];
  __shared__ bf16_t Vt[DH * VS_S];       // [d][key]
  __shared__ bf16_t Ps[8 * 16 * PS_S];   // per-wave P tiles

  const int tid  = threadIdx.x;
  const int lane = tid & 31;
  const int wave = tid >> 5;
  const int lo   = lane & 15;
  const int hi   = lane >> 4;

  const int bh = blockIdx.x;
  const int b  = bh / H, h = bh % H;
  const int q0 = blockIdx.y * NQT;
  const int strideH = 3 * H * DH;        // 2304 halves per token
  const float scale = 0.125f;            // 64^-0.5

  const bf16_t* Qbase = QKV + (size_t)(b * N) * strideH + h * DH;
  const bf16_t* Kbase = Qbase + H * DH;
  const bf16_t* Vbase = Qbase + 2 * H * DH;

  // ---- Q tile: 128 x 64 halves, async copy in 16B chunks -----------------
  {
    const unsigned lds0 = (unsigned)(size_t)Qs;
#pragma unroll
    for (int i = 0; i < 4; ++i) {
      int u  = tid + i * 256;            // 0..1023
      int qi = u >> 3, c = u & 7;
      async_copy_b128(lds0 + (unsigned)(qi * QS_S + c * 8) * 2,
                      (unsigned)(((q0 + qi) * strideH + c * 8) * 2),
                      (const void*)Qbase);
    }
  }

  v8f oacc[4];
#pragma unroll
  for (int j = 0; j < 4; ++j) oacc[j] = {0.f, 0.f, 0.f, 0.f, 0.f, 0.f, 0.f, 0.f};
  float mrow[8], lrow[8];
#pragma unroll
  for (int r = 0; r < 8; ++r) { mrow[r] = -1e30f; lrow[r] = 0.0f; }

  for (int kc = 0; kc < N; kc += NKC) {
    __syncthreads();
    {                                    // K chunk: 64 x 64 halves, async
      const unsigned lds0 = (unsigned)(size_t)Ks;
#pragma unroll
      for (int i = 0; i < 2; ++i) {
        int u  = tid + i * 256;          // 0..511
        int ki = u >> 3, c = u & 7;
        async_copy_b128(lds0 + (unsigned)(ki * KS_S + c * 8) * 2,
                        (unsigned)(((kc + ki) * strideH + c * 8) * 2),
                        (const void*)Kbase);
      }
    }
#pragma unroll
    for (int i = 0; i < 2; ++i) {        // V chunk, transposed into LDS
      int u  = tid + i * 256;
      int ki = u >> 3, c = u & 7;
      uint4 raw = *(const uint4*)(Vbase + (size_t)(kc + ki) * strideH + c * 8);
      const bf16_t* hs = (const bf16_t*)&raw;
#pragma unroll
      for (int j = 0; j < 8; ++j) Vt[(c * 8 + j) * VS_S + ki] = hs[j];
    }
    wait_async0();
    __syncthreads();

    // ---- S = Q @ K^T (16 q x 64 k per wave; D=64 -> 2 WMMA k-steps) ------
    Frag16 aq[2];
    {
      const unsigned int* row = (const unsigned int*)&Qs[(wave * 16 + lo) * QS_S];
#pragma unroll
      for (int ks = 0; ks < 2; ++ks)
#pragma unroll
        for (int e = 0; e < 4; ++e) {
          aq[ks].u[e]     = row[16 * ks + 8 * hi + e];
          aq[ks].u[4 + e] = row[16 * ks + 8 * hi + 4 + e];
        }
    }
    v8f sfrag[4];
#pragma unroll
    for (int ni = 0; ni < 4; ++ni) {
      const unsigned int* col = (const unsigned int*)&Ks[(ni * 16 + lo) * KS_S];
      Frag16 bk0, bk1;
#pragma unroll
      for (int e = 0; e < 8; ++e) {
        bk0.u[e] = col[8 * hi + e];
        bk1.u[e] = col[16 + 8 * hi + e];
      }
      v8f z = {0.f, 0.f, 0.f, 0.f, 0.f, 0.f, 0.f, 0.f};
      z = wmma_bf16(aq[0].v, bk0.v, z);
      z = wmma_bf16(aq[1].v, bk1.v, z);
      sfrag[ni] = z * scale;             // softmax scale folded here
    }

    // ---- online softmax (row = r + 8*hi; lanes `lo` span keys) -----------
#pragma unroll
    for (int r = 0; r < 8; ++r) {
      float mx = -1e30f;
#pragma unroll
      for (int ni = 0; ni < 4; ++ni) mx = fmaxf(mx, sfrag[ni][r]);
#pragma unroll
      for (int m = 1; m < 16; m <<= 1) mx = fmaxf(mx, __shfl_xor(mx, m, 32));
      const float mnew  = fmaxf(mrow[r], mx);
      const float alpha = __expf(mrow[r] - mnew);
      mrow[r] = mnew;
      lrow[r] *= alpha;
#pragma unroll
      for (int j = 0; j < 4; ++j) oacc[j][r] *= alpha;
      float ps = 0.0f;
#pragma unroll
      for (int ni = 0; ni < 4; ++ni) {
        float p = __expf(sfrag[ni][r] - mnew);
        sfrag[ni][r] = p;
        ps += p;
      }
#pragma unroll
      for (int m = 1; m < 16; m <<= 1) ps += __shfl_xor(ps, m, 32);
      lrow[r] += ps;
    }

    // ---- P through per-wave LDS into A layout, then O += P @ V -----------
    bf16_t* Pw = &Ps[wave * 16 * PS_S];
#pragma unroll
    for (int ni = 0; ni < 4; ++ni)
#pragma unroll
      for (int r = 0; r < 8; ++r)
        Pw[(r + 8 * hi) * PS_S + ni * 16 + lo] = f2bf(sfrag[ni][r]);
    // same-wave LDS ops are in-order: no barrier needed before re-reading Pw

    const unsigned int* prow = (const unsigned int*)&Pw[lo * PS_S];
#pragma unroll
    for (int ks = 0; ks < 2; ++ks) {     // 64 keys -> 2 WMMA k-steps
      Frag16 ap;
#pragma unroll
      for (int e = 0; e < 4; ++e) {
        ap.u[e]     = prow[16 * ks + 8 * hi + e];
        ap.u[4 + e] = prow[16 * ks + 8 * hi + 4 + e];
      }
#pragma unroll
      for (int j = 0; j < 4; ++j) {
        const unsigned int* vcol = (const unsigned int*)&Vt[(j * 16 + lo) * VS_S];
        Frag16 bv;
#pragma unroll
        for (int e = 0; e < 8; ++e) bv.u[e] = vcol[16 * ks + 8 * hi + e];
        oacc[j] = wmma_bf16(ap.v, bv.v, oacc[j]);
      }
    }
  }

  // ---- epilogue: out[b, n, h*64 + d] = O / l (bf16) ----------------------
  const size_t base = (size_t)(b * N) * (H * DH);
#pragma unroll
  for (int j = 0; j < 4; ++j) {
    const int col = h * DH + j * 16 + lo;
#pragma unroll
    for (int r = 0; r < 8; ++r) {
      const int qi = q0 + wave * 16 + r + 8 * hi;
      O[base + (size_t)qi * (H * DH) + col] = f2bf(oacc[j][r] / lrow[r]);
    }
  }
}

// ---------------------------------------------------------------------------
// Host orchestration: 12-layer scan, all on `stream`.
// ---------------------------------------------------------------------------
extern "C" void kernel_launch(void* const* d_in, const int* in_sizes, int n_in,
                              void* d_out, int out_size, void* d_ws, size_t ws_size,
                              hipStream_t stream) {
  (void)in_sizes; (void)n_in; (void)out_size; (void)ws_size;

  const int B = 8, N = 1024, EMB = 768, H = 12, FF = 3072, DEPTH = 12;
  const int M = B * N;  // 8192 token rows

  const float* x_in  = (const float*)d_in[0];
  const float* ln1w  = (const float*)d_in[1];
  const float* ln1b  = (const float*)d_in[2];
  const float* qkvw  = (const float*)d_in[3];
  const float* qkvb  = (const float*)d_in[4];
  const float* projw = (const float*)d_in[5];
  const float* projb = (const float*)d_in[6];
  const float* ln2w  = (const float*)d_in[7];
  const float* ln2b  = (const float*)d_in[8];
  const float* fc1w  = (const float*)d_in[9];
  const float* fc1b  = (const float*)d_in[10];
  const float* fc2w  = (const float*)d_in[11];
  const float* fc2b  = (const float*)d_in[12];

  char* ws = (char*)d_ws;
  size_t off = 0;
  float*  X   = (float*)(ws + off);  off += (size_t)M * EMB * 4;      // 25 MB f32
  bf16_t* Hb  = (bf16_t*)(ws + off); off += (size_t)M * EMB * 2;      // 13 MB bf16
  bf16_t* QKV = (bf16_t*)(ws + off); off += (size_t)M * 3 * EMB * 2;  // 38 MB bf16
  bf16_t* ATT = (bf16_t*)(ws + off); off += (size_t)M * EMB * 2;      // 13 MB bf16
  bf16_t* FFH = (bf16_t*)(ws + off); off += (size_t)M * FF * 2;       // 50 MB bf16

  hipMemcpyAsync(X, x_in, (size_t)M * EMB * 4, hipMemcpyDeviceToDevice, stream);

  const dim3 blk(256);
  for (int l = 0; l < DEPTH; ++l) {
    // LN1 -> bf16
    layernorm_kernel<<<M / 8, blk, 0, stream>>>(X, ln1w + l * EMB, ln1b + l * EMB, Hb);
    // QKV projection -> bf16
    gemm_kernel<<<dim3(3 * EMB / GBN, M / GBM), blk, 0, stream>>>(
        Hb, qkvw + (size_t)l * EMB * 3 * EMB, qkvb + (size_t)l * 3 * EMB,
        nullptr, QKV, M, EMB, 3 * EMB, /*flags=*/2);
    // fused flash attention -> bf16
    attn_kernel<<<dim3(B * H, N / NQT), blk, 0, stream>>>(QKV, ATT, B, N, H);
    // output projection + residual -> f32 X (in-place residual is per-element)
    gemm_kernel<<<dim3(EMB / GBN, M / GBM), blk, 0, stream>>>(
        ATT, projw + (size_t)l * EMB * EMB, projb + (size_t)l * EMB,
        X, X, M, EMB, EMB, /*flags=*/0);
    // LN2 -> bf16
    layernorm_kernel<<<M / 8, blk, 0, stream>>>(X, ln2w + l * EMB, ln2b + l * EMB, Hb);
    // FC1 + exact GELU -> bf16
    gemm_kernel<<<dim3(FF / GBN, M / GBM), blk, 0, stream>>>(
        Hb, fc1w + (size_t)l * EMB * FF, fc1b + (size_t)l * FF,
        nullptr, FFH, M, EMB, FF, /*flags=*/3);
    // FC2 + residual -> f32 (last layer straight into d_out)
    float* dst = (l == DEPTH - 1) ? (float*)d_out : X;
    gemm_kernel<<<dim3(EMB / GBN, M / GBM), blk, 0, stream>>>(
        FFH, fc2w + (size_t)l * FF * EMB, fc2b + (size_t)l * EMB,
        X, dst, M, FF, EMB, /*flags=*/0);
  }
}
